// RefinedMcCarthyGPT_78451872629409
// MI455X (gfx1250) — compile-verified, matrix operand order checked by reading, not run on hardware
//
#include <hip/hip_runtime.h>
#include <hip/hip_bf16.h>

// ---------------- problem constants ----------------
#define TT     1024
#define BBATCH 2
#define EE     512
#define HH     8
#define HDM    64
#define LLAY   8
#define HIDN   1365
#define HIDP   1376    // HIDN padded to %32 (GEMM K) — zero-filled
#define NGP    1408    // HIDN padded to %64 (GEMM N tiles) — zero-filled
#define E3     1536
#define ROWS   2048    // B*T tokens
#define VV     50257
#define VP     50304   // V padded to %64 — zero-filled

// ---------------- WMMA types ----------------
typedef __attribute__((ext_vector_type(16))) __bf16 v16bf;
typedef __attribute__((ext_vector_type(8)))  float  v8f;
typedef __attribute__((ext_vector_type(4)))  int    v4i;

union FragBF { v16bf v; uint4 q[2]; unsigned short s[16]; };
union Pack8  { uint4 q; unsigned short s[8]; };

__device__ __forceinline__ unsigned short f2bf(float f) {
  union { float f; unsigned u; } c; c.f = f;
  unsigned u = c.u;
  return (unsigned short)((u + 0x7FFFu + ((u >> 16) & 1u)) >> 16);
}

// ---------------- gfx1250 async LDS DMA (guarded; sync fallback) ----------------
#if defined(__HIP_DEVICE_COMPILE__) && __has_builtin(__builtin_amdgcn_global_load_async_to_lds_b128)
#define HAVE_ASYNC 1
#else
#define HAVE_ASYNC 0
#endif

#if HAVE_ASYNC
__device__ __forceinline__ void async_cp16(const unsigned short* g, unsigned short* l) {
  __builtin_amdgcn_global_load_async_to_lds_b128(
      (__attribute__((address_space(1))) v4i*)g,
      (__attribute__((address_space(3))) v4i*)l,
      0, 0);
}
__device__ __forceinline__ void wait_async0() {
#if __has_builtin(__builtin_amdgcn_s_wait_asynccnt)
  __builtin_amdgcn_s_wait_asynccnt(0);
#else
  asm volatile("s_wait_asynccnt 0x0" ::: "memory");
#endif
}
#endif

// ---------------- weight transpose + fp32->bf16 + zero pad ----------------
// src: fp32 [K,N] row-major (per layer). dst: bf16 [Np,Kp] row-major (B^T), zero-padded.
__global__ __launch_bounds__(256) void transpose_cvt_kernel(const float* __restrict__ src,
                                                            unsigned short* __restrict__ dst,
                                                            int K, int N, int Kp, int Np) {
  __shared__ float t32[32][33];
  const float* s = src + (size_t)blockIdx.z * K * N;
  unsigned short* d = dst + (size_t)blockIdx.z * Np * Kp;
  int nb = blockIdx.x * 32, kb = blockIdx.y * 32;
  int tx = threadIdx.x & 31, ty = threadIdx.x >> 5;   // 32x8
#pragma unroll
  for (int i = 0; i < 32; i += 8) {
    int k = kb + ty + i, n = nb + tx;
    t32[ty + i][tx] = (k < K && n < N) ? s[(size_t)k * N + n] : 0.f;
  }
  __syncthreads();
#pragma unroll
  for (int i = 0; i < 32; i += 8) {
    int n = nb + ty + i, k = kb + tx;
    if (n < Np && k < Kp) d[(size_t)n * Kp + k] = f2bf(t32[tx][ty + i]);
  }
}

// W_emb fp32 [V,E] -> bf16 [VP,E], pad rows with zeros (already B^T layout for tied head)
__global__ __launch_bounds__(256) void cvt_pad_kernel(const float* __restrict__ src,
                                                      unsigned short* __restrict__ dst,
                                                      int vrows) {
  long long n = (long long)VP * EE;
  long long i = (long long)blockIdx.x * blockDim.x + threadIdx.x;
  long long stride = (long long)gridDim.x * blockDim.x;
  for (; i < n; i += stride) {
    int row = (int)(i >> 9);   // /EE
    dst[i] = (row < vrows) ? f2bf(src[i]) : (unsigned short)0;
  }
}

// ---------------- embedding: x = W_emb[idx] + pos_emb ----------------
__global__ __launch_bounds__(256) void embed_kernel(const int* __restrict__ idx,
                                                    const float* __restrict__ W,
                                                    const float* __restrict__ pos,
                                                    float* __restrict__ x) {
  int row = blockIdx.x;            // b*T + t
  int t = row & (TT - 1);
  int id = idx[row];
  for (int e = threadIdx.x; e < EE; e += 256)
    x[(size_t)row * EE + e] = W[(size_t)id * EE + e] + pos[(size_t)t * EE + e];
}

// ---------------- LayerNorm (fp32 in -> bf16 out) ----------------
__global__ __launch_bounds__(256) void ln_kernel(const float* __restrict__ x,
                                                 const float* __restrict__ sc,
                                                 const float* __restrict__ bi,
                                                 unsigned short* __restrict__ out) {
  int row = blockIdx.x, tid = threadIdx.x;
  const float* xr = x + (size_t)row * EE;
  float2 v = *(const float2*)&xr[tid * 2];
  float sum = v.x + v.y;
  float sq  = v.x * v.x + v.y * v.y;
#pragma unroll
  for (int m = 16; m >= 1; m >>= 1) {
    sum += __shfl_xor(sum, m, 32);
    sq  += __shfl_xor(sq,  m, 32);
  }
  __shared__ float red[16];
  int lane = tid & 31, w = tid >> 5;
  if (lane == 0) { red[w] = sum; red[8 + w] = sq; }
  __syncthreads();
  float ts = 0.f, tq = 0.f;
#pragma unroll
  for (int i = 0; i < 8; ++i) { ts += red[i]; tq += red[8 + i]; }
  float mean = ts * (1.f / EE);
  float var  = tq * (1.f / EE) - mean * mean;
  float rs   = rsqrtf(var + 1e-5f);
  int e0 = tid * 2;
  out[(size_t)row * EE + e0]     = f2bf((v.x - mean) * rs * sc[e0]     + bi[e0]);
  out[(size_t)row * EE + e0 + 1] = f2bf((v.y - mean) * rs * sc[e0 + 1] + bi[e0 + 1]);
}

// ---------------- SwiGLU: m = g*sigmoid(g)*u (bf16 out, K-padded to HIDP) ----------------
__global__ __launch_bounds__(256) void swiglu_kernel(const float* __restrict__ g,
                                                     const float* __restrict__ u,
                                                     unsigned short* __restrict__ out) {
  int row = blockIdx.x;
  for (int c = threadIdx.x; c < HIDP; c += 256) {
    float val = 0.f;
    if (c < HIDN) {
      float gv = g[(size_t)row * HIDN + c];
      float s  = 1.f / (1.f + __expf(-gv));
      val = gv * s * u[(size_t)row * HIDN + c];
    }
    out[(size_t)row * HIDP + c] = f2bf(val);
  }
}

// ---------------- bf16 WMMA GEMM (async-LDS double-buffered) ----------------
// C[M,N](f32) = A[M,Kp](bf16) * Bt[Np,Kp](bf16, pre-transposed/padded)  (+R if RES)
// Kp % 32 == 0; grid.x covers Np/64 tiles (B rows always in-bounds); only C store is guarded.
#define LDS_STR 40   // padded LDS row stride (bf16): 80B -> conflict-free ds_load_b128

template <bool RES>
__global__ __launch_bounds__(256) void gemm_bf16(const unsigned short* __restrict__ A,
                                                 const unsigned short* __restrict__ Bt,
                                                 const float* __restrict__ R,
                                                 float* __restrict__ C,
                                                 int M, int N, int Kp) {
  __shared__ __align__(16) unsigned short sA[2][128 * LDS_STR];
  __shared__ __align__(16) unsigned short sB[2][64 * LDS_STR];

  int tid = threadIdx.x;
  int lane = tid & 31, wave = tid >> 5;
  int hi = lane >> 4, lr = lane & 15;
  int wm = wave >> 1, wn = wave & 1;
  int m0 = blockIdx.y * 128;
  int nb = blockIdx.x * 64;

  int r = tid >> 2, q = (tid & 3) * 8;   // copy lane mapping: 16B chunks

  v8f acc[2][2];
#pragma unroll
  for (int i = 0; i < 2; ++i)
#pragma unroll
    for (int j = 0; j < 2; ++j) acc[i][j] = (v8f){0.f,0.f,0.f,0.f,0.f,0.f,0.f,0.f};

  auto stage = [&](int buf, int k0) {
#if HAVE_ASYNC
    async_cp16(A + (size_t)(m0 + r) * Kp + k0 + q,       &sA[buf][r * LDS_STR + q]);
    async_cp16(A + (size_t)(m0 + r + 64) * Kp + k0 + q,  &sA[buf][(r + 64) * LDS_STR + q]);
    async_cp16(Bt + (size_t)(nb + r) * Kp + k0 + q,      &sB[buf][r * LDS_STR + q]);
#else
    *(uint4*)&sA[buf][r * LDS_STR + q] =
        *(const uint4*)(A + (size_t)(m0 + r) * Kp + k0 + q);
    *(uint4*)&sA[buf][(r + 64) * LDS_STR + q] =
        *(const uint4*)(A + (size_t)(m0 + r + 64) * Kp + k0 + q);
    *(uint4*)&sB[buf][r * LDS_STR + q] =
        *(const uint4*)(Bt + (size_t)(nb + r) * Kp + k0 + q);
#endif
  };

  int nkt = Kp >> 5;
  stage(0, 0);
#if HAVE_ASYNC
  wait_async0();
#endif
  __syncthreads();

  for (int kt = 0; kt < nkt; ++kt) {
    int cur = kt & 1;
    if (kt + 1 < nkt) stage(cur ^ 1, (kt + 1) * 32);   // DMA overlaps WMMA below

    FragBF a[2], b[2];
#pragma unroll
    for (int mi = 0; mi < 2; ++mi) {
      const unsigned short* p = &sA[cur][(wm * 32 + mi * 16 + lr) * LDS_STR + 8 * hi];
      a[mi].q[0] = *(const uint4*)p;
      a[mi].q[1] = *(const uint4*)(p + 16);
    }
#pragma unroll
    for (int ni = 0; ni < 2; ++ni) {
      const unsigned short* p = &sB[cur][(wn * 32 + ni * 16 + lr) * LDS_STR + 8 * hi];
      b[ni].q[0] = *(const uint4*)p;
      b[ni].q[1] = *(const uint4*)(p + 16);
    }
#pragma unroll
    for (int mi = 0; mi < 2; ++mi)
#pragma unroll
      for (int ni = 0; ni < 2; ++ni)
        acc[mi][ni] = __builtin_amdgcn_wmma_f32_16x16x32_bf16(
            false, a[mi].v, false, b[ni].v, (short)0, acc[mi][ni], false, false);

#if HAVE_ASYNC
    wait_async0();
#endif
    __syncthreads();
  }

#pragma unroll
  for (int mi = 0; mi < 2; ++mi)
#pragma unroll
    for (int ni = 0; ni < 2; ++ni)
#pragma unroll
      for (int rr = 0; rr < 8; ++rr) {
        int row = m0 + wm * 32 + mi * 16 + 8 * hi + rr;
        int col = nb + wn * 32 + ni * 16 + lr;
        if (col < N) {
          float v = acc[mi][ni][rr];
          if (RES) v += R[(size_t)row * N + col];
          C[(size_t)row * N + col] = v;
        }
      }
}

// ---------------- Flash attention (WMMA QK^T and PV, online softmax) ----------------
// qkv: fp32 [2048, 1536] (q|k|v). out: bf16 [2048, 512].
// grid (T/128, H, B); block 256 = 8 waves; each wave owns 16 queries.
__global__ __launch_bounds__(256) void attn_kernel(const float* __restrict__ qkv,
                                                   unsigned short* __restrict__ out) {
  __shared__ __align__(16) unsigned short sK[32 * 72];          // K tile  [kk][d]
  __shared__ __align__(16) unsigned short sVT[64 * LDS_STR];    // V^T tile [d][kk]
  __shared__ __align__(16) unsigned short sP[8 * 16 * LDS_STR]; // P per wave [q][kk]

  int tid = threadIdx.x, lane = tid & 31, wave = tid >> 5;
  int hi = lane >> 4, lr = lane & 15;
  int qb = blockIdx.x, h = blockIdx.y, b = blockIdx.z;
  int qbase = qb * 128 + wave * 16;
  float slope = exp2f(-(float)(h + 1));   // 1 / 2^(8*(h+1)/H)

  // Q fragments (pre-scaled by 1/sqrt(HD)); A-layout: lane row = lr, K(j) mapping
  FragBF qf[2];
  {
    int qrow = qbase + lr;
    const float* qp = qkv + (size_t)(b * TT + qrow) * E3 + h * HDM;
#pragma unroll
    for (int g2 = 0; g2 < 2; ++g2) {
      const float* p0 = qp + g2 * 32 + 8 * hi;
      const float* p1 = p0 + 16;
      float4 a0 = ((const float4*)p0)[0], a1 = ((const float4*)p0)[1];
      float4 b0 = ((const float4*)p1)[0], b1 = ((const float4*)p1)[1];
      float f[16] = {a0.x,a0.y,a0.z,a0.w, a1.x,a1.y,a1.z,a1.w,
                     b0.x,b0.y,b0.z,b0.w, b1.x,b1.y,b1.z,b1.w};
#pragma unroll
      for (int j = 0; j < 16; ++j) qf[g2].s[j] = f2bf(f[j] * 0.125f);
    }
  }

  float m_r[8], l_r[8];
  v8f o[4];
#pragma unroll
  for (int rr = 0; rr < 8; ++rr) { m_r[rr] = -1e30f; l_r[rr] = 0.f; }
#pragma unroll
  for (int gd = 0; gd < 4; ++gd) o[gd] = (v8f){0.f,0.f,0.f,0.f,0.f,0.f,0.f,0.f};

  int nkt = 4 * (qb + 1);   // uniform across block -> __syncthreads legal
  for (int kt = 0; kt < nkt; ++kt) {
    int kbase = kt * 32;
    __syncthreads();
    // cooperative load: K tile [32k x 64d], V tile transposed [64d x 32k]
    {
      int kk = tid >> 3, d0 = (tid & 7) * 8;
      const float* kp = qkv + (size_t)(b * TT + kbase + kk) * E3 + EE + h * HDM + d0;
      float4 a0 = ((const float4*)kp)[0], a1 = ((const float4*)kp)[1];
      Pack8 pk;
      pk.s[0]=f2bf(a0.x); pk.s[1]=f2bf(a0.y); pk.s[2]=f2bf(a0.z); pk.s[3]=f2bf(a0.w);
      pk.s[4]=f2bf(a1.x); pk.s[5]=f2bf(a1.y); pk.s[6]=f2bf(a1.z); pk.s[7]=f2bf(a1.w);
      *(uint4*)&sK[kk * 72 + d0] = pk.q;
      const float* vp = kp + EE;
      float4 c0 = ((const float4*)vp)[0], c1 = ((const float4*)vp)[1];
      float vv[8] = {c0.x,c0.y,c0.z,c0.w, c1.x,c1.y,c1.z,c1.w};
#pragma unroll
      for (int i = 0; i < 8; ++i) sVT[(d0 + i) * LDS_STR + kk] = f2bf(vv[i]);
    }
    __syncthreads();

    bool active = (kbase <= qbase + 15);   // wave-uniform
    float alpha[8];
    if (active) {
      v8f sg[2];
#pragma unroll
      for (int g2 = 0; g2 < 2; ++g2) {
        v8f s = (v8f){0.f,0.f,0.f,0.f,0.f,0.f,0.f,0.f};
#pragma unroll
        for (int dg = 0; dg < 2; ++dg) {
          FragBF kf;
          const unsigned short* p = &sK[(g2 * 16 + lr) * 72 + dg * 32 + 8 * hi];
          kf.q[0] = *(const uint4*)p;
          kf.q[1] = *(const uint4*)(p + 16);
          s = __builtin_amdgcn_wmma_f32_16x16x32_bf16(
              false, qf[dg].v, false, kf.v, (short)0, s, false, false);
        }
        sg[g2] = s;
      }
      // ALiBi (+slope*(q-k), faithful to ref) + causal mask, all in float
      float fb0 = (float)(qbase + 8 * hi - kbase - lr);
      float fb1 = fb0 - 16.f;
      float rmax[8];
#pragma unroll
      for (int rr = 0; rr < 8; ++rr) {
        float d0 = fb0 + (float)rr;
        float d1 = fb1 + (float)rr;
        float v0 = fmaf(slope, d0, sg[0][rr]);
        float v1 = fmaf(slope, d1, sg[1][rr]);
        v0 = (d0 < 0.f) ? -1e30f : v0;
        v1 = (d1 < 0.f) ? -1e30f : v1;
        sg[0][rr] = v0; sg[1][rr] = v1;
        rmax[rr] = fmaxf(v0, v1);
      }
#pragma unroll
      for (int msk = 8; msk >= 1; msk >>= 1)
#pragma unroll
        for (int rr = 0; rr < 8; ++rr)
          rmax[rr] = fmaxf(rmax[rr], __shfl_xor(rmax[rr], msk, 32));

      float rsum[8];
      int wb = wave * 16 * LDS_STR;
#pragma unroll
      for (int rr = 0; rr < 8; ++rr) {
        float mn = fmaxf(m_r[rr], rmax[rr]);
        alpha[rr] = __expf(m_r[rr] - mn);
        float p0 = __expf(sg[0][rr] - mn);
        float p1 = __expf(sg[1][rr] - mn);
        m_r[rr] = mn;
        rsum[rr] = p0 + p1;
        int prow = 8 * hi + rr;
        sP[wb + prow * LDS_STR + lr]      = f2bf(p0);
        sP[wb + prow * LDS_STR + 16 + lr] = f2bf(p1);
      }
#pragma unroll
      for (int msk = 8; msk >= 1; msk >>= 1)
#pragma unroll
        for (int rr = 0; rr < 8; ++rr)
          rsum[rr] += __shfl_xor(rsum[rr], msk, 32);
#pragma unroll
      for (int rr = 0; rr < 8; ++rr) l_r[rr] = l_r[rr] * alpha[rr] + rsum[rr];
    }
    __syncthreads();   // order P writes before fragment reads
    if (active) {
      FragBF pf;
      const unsigned short* pb = &sP[wave * 16 * LDS_STR + lr * LDS_STR + 8 * hi];
      pf.q[0] = *(const uint4*)pb;
      pf.q[1] = *(const uint4*)(pb + 16);
#pragma unroll
      for (int gd = 0; gd < 4; ++gd) {
        FragBF vf;
        const unsigned short* vb = &sVT[(gd * 16 + lr) * LDS_STR + 8 * hi];
        vf.q[0] = *(const uint4*)vb;
        vf.q[1] = *(const uint4*)(vb + 16);
        v8f t = o[gd];
#pragma unroll
        for (int rr = 0; rr < 8; ++rr) t[rr] *= alpha[rr];
        o[gd] = __builtin_amdgcn_wmma_f32_16x16x32_bf16(
            false, pf.v, false, vf.v, (short)0, t, false, false);
      }
    }
  }

#pragma unroll
  for (int gd = 0; gd < 4; ++gd)
#pragma unroll
    for (int rr = 0; rr < 8; ++rr) {
      int qq = qbase + 8 * hi + rr;
      float val = o[gd][rr] / l_r[rr];
      out[(size_t)(b * TT + qq) * EE + h * HDM + gd * 16 + lr] = f2bf(val);
    }
}

// ---------------- host launcher ----------------
extern "C" void kernel_launch(void* const* d_in, const int* in_sizes, int n_in,
                              void* d_out, int out_size, void* d_ws, size_t ws_size,
                              hipStream_t stream) {
  const int*   idx    = (const int*)  d_in[0];
  const float* W_emb  = (const float*)d_in[1];
  const float* pos_emb= (const float*)d_in[2];
  const float* ln1_s  = (const float*)d_in[3];
  const float* ln1_b  = (const float*)d_in[4];
  const float* qkv_w  = (const float*)d_in[5];
  const float* proj_w = (const float*)d_in[6];
  const float* ln2_s  = (const float*)d_in[7];
  const float* ln2_b  = (const float*)d_in[8];
  const float* gate_w = (const float*)d_in[9];
  const float* up_w   = (const float*)d_in[10];
  const float* down_w = (const float*)d_in[11];
  const float* lnf_s  = (const float*)d_in[12];
  const float* lnf_b  = (const float*)d_in[13];
  float* logits = (float*)d_out;

  char* wp = (char*)d_ws;
  auto carve = [&](size_t bytes) -> void* {
    void* p = (void*)wp;
    wp += (bytes + 255) & ~(size_t)255;
    return p;
  };
  // pre-transposed, padded bf16 weights (B^T layout [Np, Kp])
  unsigned short* wqkvT  = (unsigned short*)carve((size_t)LLAY * E3  * EE   * 2); // [1536,512] x8
  unsigned short* wprojT = (unsigned short*)carve((size_t)LLAY * EE  * EE   * 2); // [512,512]  x8
  unsigned short* wgateT = (unsigned short*)carve((size_t)LLAY * NGP * EE   * 2); // [1408,512] x8
  unsigned short* wupT   = (unsigned short*)carve((size_t)LLAY * NGP * EE   * 2);
  unsigned short* wdownT = (unsigned short*)carve((size_t)LLAY * EE  * HIDP * 2); // [512,1376] x8
  unsigned short* wembB  = (unsigned short*)carve((size_t)VP * EE * 2);           // [50304,512]
  float*          x      = (float*)carve((size_t)ROWS * EE * 4);
  unsigned short* hbuf   = (unsigned short*)carve((size_t)ROWS * EE * 2);
  float*          qkvb   = (float*)carve((size_t)ROWS * E3 * 4);
  unsigned short* attnb  = (unsigned short*)carve((size_t)ROWS * EE * 2);
  float*          gbuf   = (float*)carve((size_t)ROWS * HIDN * 4);
  float*          ubuf   = (float*)carve((size_t)ROWS * HIDN * 4);
  unsigned short* mbuf   = (unsigned short*)carve((size_t)ROWS * HIDP * 2);

  // 1) weight prep: transpose + bf16 + pad (one-time; then weights live in 192MB L2)
  transpose_cvt_kernel<<<dim3(E3 / 32, EE / 32, LLAY), 256, 0, stream>>>(
      qkv_w, wqkvT, EE, E3, EE, E3);
  transpose_cvt_kernel<<<dim3(EE / 32, EE / 32, LLAY), 256, 0, stream>>>(
      proj_w, wprojT, EE, EE, EE, EE);
  transpose_cvt_kernel<<<dim3(NGP / 32, EE / 32, LLAY), 256, 0, stream>>>(
      gate_w, wgateT, EE, HIDN, EE, NGP);
  transpose_cvt_kernel<<<dim3(NGP / 32, EE / 32, LLAY), 256, 0, stream>>>(
      up_w, wupT, EE, HIDN, EE, NGP);
  transpose_cvt_kernel<<<dim3(EE / 32, HIDP / 32, LLAY), 256, 0, stream>>>(
      down_w, wdownT, HIDN, EE, HIDP, EE);
  cvt_pad_kernel<<<4096, 256, 0, stream>>>(W_emb, wembB, VV);

  // 2) embedding
  embed_kernel<<<ROWS, 256, 0, stream>>>(idx, W_emb, pos_emb, x);

  // 3) transformer blocks
  for (int l = 0; l < LLAY; ++l) {
    ln_kernel<<<ROWS, 256, 0, stream>>>(x, ln1_s + (size_t)l * EE, ln1_b + (size_t)l * EE, hbuf);
    gemm_bf16<false><<<dim3(E3 / 64, ROWS / 128), 256, 0, stream>>>(
        hbuf, wqkvT + (size_t)l * E3 * EE, nullptr, qkvb, ROWS, E3, EE);
    attn_kernel<<<dim3(TT / 128, HH, BBATCH), 256, 0, stream>>>(qkvb, attnb);
    gemm_bf16<true><<<dim3(EE / 64, ROWS / 128), 256, 0, stream>>>(
        attnb, wprojT + (size_t)l * EE * EE, x, x, ROWS, EE, EE);
    ln_kernel<<<ROWS, 256, 0, stream>>>(x, ln2_s + (size_t)l * EE, ln2_b + (size_t)l * EE, hbuf);
    gemm_bf16<false><<<dim3(NGP / 64, ROWS / 128), 256, 0, stream>>>(
        hbuf, wgateT + (size_t)l * NGP * EE, nullptr, gbuf, ROWS, HIDN, EE);
    gemm_bf16<false><<<dim3(NGP / 64, ROWS / 128), 256, 0, stream>>>(
        hbuf, wupT + (size_t)l * NGP * EE, nullptr, ubuf, ROWS, HIDN, EE);
    swiglu_kernel<<<ROWS, 256, 0, stream>>>(gbuf, ubuf, mbuf);
    gemm_bf16<true><<<dim3(EE / 64, ROWS / 128), 256, 0, stream>>>(
        mbuf, wdownT + (size_t)l * EE * HIDP, x, x, ROWS, EE, HIDP);
  }

  // 4) final LN + tied head: logits = h @ W_emb^T (Bt = padded bf16 W_emb [VP,512])
  ln_kernel<<<ROWS, 256, 0, stream>>>(x, lnf_s, lnf_b, hbuf);
  gemm_bf16<false><<<dim3(VP / 64, ROWS / 128), 256, 0, stream>>>(
      hbuf, wembB, nullptr, logits, ROWS, VV, EE);
}
